// GraphMultiHeadAttention_6141803233661
// MI455X (gfx1250) — compile-verified, matrix operand order checked by reading, not run on hardware
//
#include <hip/hip_runtime.h>
#include <hip/hip_bf16.h>
#include <math.h>

// Problem constants (fixed by the reference).
#define B_  64
#define N_  512
#define D_  256
#define H_  8
#define DK_ 32

typedef __attribute__((ext_vector_type(16))) _Float16 v16h;
typedef __attribute__((ext_vector_type(8)))  _Float16 v8h;
typedef __attribute__((ext_vector_type(8)))  float    v8f;

union V16H { v16h v; v8h h[2]; };

// Load a 16x32 f16 WMMA fragment (A-style: lane = row M, halves = K) from a
// row-major matrix. Also used for B-fragments by pointing `base` at a
// column-major (N-major) matrix so lane = column N, halves = K.
// CDNA5 16-bit A layout: lanes 0-15 halves 0..7 -> K 0..7, halves 8..15 -> K 16..23;
// lanes 16-31 get K+8 on both groups. Two 16B loads per lane.
__device__ __forceinline__ v16h load_frag16(const _Float16* __restrict__ base,
                                            int stride, int row0, int k0) {
  const int lane = threadIdx.x & 31;
  const int r  = row0 + (lane & 15);
  const int kb = k0 + ((lane >> 4) << 3);
  const _Float16* p = base + (size_t)r * stride + kb;
  V16H u;
  u.h[0] = *(const v8h*)(p);        // K = kb .. kb+7
  u.h[1] = *(const v8h*)(p + 16);   // K = kb+16 .. kb+23
  return u.v;
}

// ---------------- prep kernels ----------------

__global__ void prep_h_kernel(const float* __restrict__ h, _Float16* __restrict__ h16, int n) {
  int i = blockIdx.x * blockDim.x + threadIdx.x;
  if (i < n) h16[i] = (_Float16)h[i];
}

// Transpose + convert the 4 weight matrices: wt[w][n][k] = (f16) W[w][k][n]
__global__ void prep_w_kernel(const float* __restrict__ Wq, const float* __restrict__ Wk,
                              const float* __restrict__ Wv, const float* __restrict__ Wo,
                              _Float16* __restrict__ wt) {
  int i = blockIdx.x * blockDim.x + threadIdx.x;      // 4 * 256 * 256
  int w = i >> 16;
  int rem = i & 65535;
  int r = rem >> 8;      // k (input dim)
  int c = rem & 255;     // n (output dim)
  const float* W = (w == 0) ? Wq : (w == 1) ? Wk : (w == 2) ? Wv : Wo;
  wt[(size_t)w * (D_ * D_) + (size_t)c * D_ + r] = (_Float16)W[(size_t)r * D_ + c];
}

// invalid[b*N+n] = any(mask[b,n,:]) || n >= node_count[b]
__global__ void prep_inv_kernel(const int* __restrict__ mask, const int* __restrict__ nc,
                                unsigned char* __restrict__ inv) {
  int i = blockIdx.x * blockDim.x + threadIdx.x;      // B * N
  int b = i >> 9;
  int n = i & (N_ - 1);
  int m = mask[2 * i] | mask[2 * i + 1];
  inv[i] = ((m != 0) || (n >= nc[b])) ? (unsigned char)1 : (unsigned char)0;
}

// ---------------- QKV projection ----------------
// One wave computes a 16x64 tile of (h @ W + b): one A-fragment is reused by
// four B-fragments per K-step (4 WMMAs / 10 b128 loads), 8 K-steps over D=256.
__global__ __launch_bounds__(32) void qkv_kernel(
    const _Float16* __restrict__ h16, const _Float16* __restrict__ wt,
    const float* __restrict__ bq, const float* __restrict__ bk, const float* __restrict__ bv,
    _Float16* __restrict__ q16, _Float16* __restrict__ k16, _Float16* __restrict__ vt16) {
  const int mt   = blockIdx.x & 31;     // n-tile within batch
  const int b    = blockIdx.x >> 5;
  const int ct   = blockIdx.y;          // 64-wide output-column group
  const int proj = blockIdx.z;          // 0=Q 1=K 2=V
  const _Float16* A  = h16 + (size_t)b * N_ * D_;
  const _Float16* Wt = wt + (size_t)proj * D_ * D_;
  v8f acc[4] = {{}, {}, {}, {}};
  for (int k0 = 0; k0 < D_; k0 += 32) {
    v16h a = load_frag16(A, D_, mt * 16, k0);
#pragma unroll
    for (int s = 0; s < 4; ++s) {
      v16h w = load_frag16(Wt, D_, ct * 64 + s * 16, k0);
      acc[s] = __builtin_amdgcn_wmma_f32_16x16x32_f16(false, a, false, w, (short)0, acc[s], false, false);
    }
  }
  const int lane  = threadIdx.x & 31;
  const int lcol  = lane & 15;
  const int rbase = (lane >> 4) << 3;
  const float* bias = (proj == 0) ? bq : (proj == 1) ? bk : bv;
#pragma unroll
  for (int s = 0; s < 4; ++s) {
    const int c    = ct * 64 + s * 16 + lcol;
    const float bb = bias[c];
    const int head = c >> 5;
    const int dk   = c & (DK_ - 1);
    for (int j = 0; j < 8; ++j) {
      const int n = mt * 16 + rbase + j;
      _Float16 v = (_Float16)(acc[s][j] + bb);
      if (proj == 0)      q16[(((size_t)b * H_ + head) * N_ + n) * DK_ + dk] = v;
      else if (proj == 1) k16[(((size_t)b * H_ + head) * N_ + n) * DK_ + dk] = v;
      else                vt16[(((size_t)b * H_ + head) * DK_ + dk) * N_ + n] = v; // V^T: d-major
    }
  }
}

// ---------------- fused flash attention ----------------
// One wave owns 16 query rows of one (batch, head). Streams keys in blocks of
// 64: 4 WMMAs for scores, one online-softmax pass (amortized over 64 keys),
// P through a 2KB LDS tile, 4 WMMAs for P@V (two K-steps of 32).
__global__ __launch_bounds__(32) void attn_kernel(
    const _Float16* __restrict__ q16, const _Float16* __restrict__ k16,
    const _Float16* __restrict__ vt16, const unsigned char* __restrict__ inv,
    _Float16* __restrict__ att16) {
  __shared__ __align__(16) _Float16 Pl[16 * 64];   // P tile, row-major over k
  const int gid  = blockIdx.x;
  const int qt   = gid & 31;
  const int head = (gid >> 5) & (H_ - 1);
  const int b    = gid >> 8;
  const _Float16* Q  = q16  + ((size_t)b * H_ + head) * N_ * DK_;
  const _Float16* K  = k16  + ((size_t)b * H_ + head) * N_ * DK_;
  const _Float16* Vt = vt16 + ((size_t)b * H_ + head) * DK_ * N_;
  const unsigned char* ib = inv + (size_t)b * N_;

  const int lane  = threadIdx.x & 31;
  const int lcol  = lane & 15;
  const int rbase = (lane >> 4) << 3;

  const v16h qa = load_frag16(Q, DK_, qt * 16, 0);   // reused for all key blocks

  bool qinv[8];
  for (int j = 0; j < 8; ++j) qinv[j] = ib[qt * 16 + rbase + j] != 0;

  const float scale = 0.17677669529663687f;          // 1/sqrt(DK)
  float m_i[8], l_i[8];
  for (int j = 0; j < 8; ++j) { m_i[j] = -3.0e38f; l_i[j] = 0.0f; }
  v8f o0 = {}, o1 = {};

  for (int kb = 0; kb < N_; kb += 64) {
    if (kb + 64 < N_) {
      __builtin_prefetch(K  + (size_t)(kb + 64) * DK_, 0, 3);   // global_prefetch_b8
      __builtin_prefetch(Vt + (size_t)(kb + 64), 0, 3);
    }
    v8f sf[4];
    bool kinv[4];
    v8f z = {};
#pragma unroll
    for (int s = 0; s < 4; ++s) {
      v16h kf = load_frag16(K, DK_, kb + s * 16, 0);
      sf[s] = __builtin_amdgcn_wmma_f32_16x16x32_f16(false, qa, false, kf, (short)0, z, false, false);
      kinv[s] = ib[kb + s * 16 + lcol] != 0;
    }
    for (int j = 0; j < 8; ++j) {
      float v[4];
#pragma unroll
      for (int s = 0; s < 4; ++s)
        v[s] = (qinv[j] || kinv[s]) ? -1.0e9f : sf[s][j] * scale;
      float mx = fmaxf(fmaxf(v[0], v[1]), fmaxf(v[2], v[3]));
      // row reduction over the 16 lanes holding this row (xor<16 stays in-group)
      mx = fmaxf(mx, __shfl_xor(mx, 1));
      mx = fmaxf(mx, __shfl_xor(mx, 2));
      mx = fmaxf(mx, __shfl_xor(mx, 4));
      mx = fmaxf(mx, __shfl_xor(mx, 8));
      float newm  = fmaxf(m_i[j], mx);
      float alpha = __expf(m_i[j] - newm);
      float p[4], rs = 0.0f;
#pragma unroll
      for (int s = 0; s < 4; ++s) { p[s] = __expf(v[s] - newm); rs += p[s]; }
      rs += __shfl_xor(rs, 1);
      rs += __shfl_xor(rs, 2);
      rs += __shfl_xor(rs, 4);
      rs += __shfl_xor(rs, 8);
      l_i[j] = l_i[j] * alpha + rs;
      m_i[j] = newm;
      o0[j] *= alpha;
      o1[j] *= alpha;
      // C-layout -> LDS (row-major in k) for the A-fragments of P@V
#pragma unroll
      for (int s = 0; s < 4; ++s)
        Pl[(rbase + j) * 64 + s * 16 + lcol] = (_Float16)p[s];
    }
    __syncthreads();   // single-wave WG: lowers to S_NOP, LDS order kept by DScnt
    v16h pf0, pf1;     // A-fragments of P: k = 0..31 and k = 32..63
    {
      const _Float16* p0 = &Pl[(lane & 15) * 64 + ((lane >> 4) << 3)];
      V16H u;
      u.h[0] = *(const v8h*)(p0);
      u.h[1] = *(const v8h*)(p0 + 16);
      pf0 = u.v;
      u.h[0] = *(const v8h*)(p0 + 32);
      u.h[1] = *(const v8h*)(p0 + 48);
      pf1 = u.v;
    }
    v16h vf00 = load_frag16(Vt, N_, 0,  kb);       // d 0..15,  k 0..31
    v16h vf01 = load_frag16(Vt, N_, 0,  kb + 32);  // d 0..15,  k 32..63
    v16h vf10 = load_frag16(Vt, N_, 16, kb);       // d 16..31, k 0..31
    v16h vf11 = load_frag16(Vt, N_, 16, kb + 32);  // d 16..31, k 32..63
    o0 = __builtin_amdgcn_wmma_f32_16x16x32_f16(false, pf0, false, vf00, (short)0, o0, false, false);
    o0 = __builtin_amdgcn_wmma_f32_16x16x32_f16(false, pf1, false, vf01, (short)0, o0, false, false);
    o1 = __builtin_amdgcn_wmma_f32_16x16x32_f16(false, pf0, false, vf10, (short)0, o1, false, false);
    o1 = __builtin_amdgcn_wmma_f32_16x16x32_f16(false, pf1, false, vf11, (short)0, o1, false, false);
    __syncthreads();
  }

  for (int j = 0; j < 8; ++j) {
    const float r = 1.0f / l_i[j];
    const int n = qt * 16 + rbase + j;
    _Float16* dst = att16 + ((size_t)b * N_ + n) * D_ + head * DK_;
    dst[lcol]      = (_Float16)(o0[j] * r);
    dst[16 + lcol] = (_Float16)(o1[j] * r);
  }
}

// ---------------- output projection + residual ----------------
// Same 16x64-tile blocking as qkv_kernel; adds bias + residual, writes f32.
__global__ __launch_bounds__(32) void oproj_kernel(
    const _Float16* __restrict__ att16, const _Float16* __restrict__ wt,
    const float* __restrict__ bo, const float* __restrict__ h, float* __restrict__ out) {
  const int mt = blockIdx.x & 31;
  const int b  = blockIdx.x >> 5;
  const int ct = blockIdx.y;            // 64-wide output-column group
  const _Float16* A  = att16 + (size_t)b * N_ * D_;
  const _Float16* Wt = wt + (size_t)3 * D_ * D_;   // Wo^T
  v8f acc[4] = {{}, {}, {}, {}};
  for (int k0 = 0; k0 < D_; k0 += 32) {
    v16h a = load_frag16(A, D_, mt * 16, k0);
#pragma unroll
    for (int s = 0; s < 4; ++s) {
      v16h w = load_frag16(Wt, D_, ct * 64 + s * 16, k0);
      acc[s] = __builtin_amdgcn_wmma_f32_16x16x32_f16(false, a, false, w, (short)0, acc[s], false, false);
    }
  }
  const int lane  = threadIdx.x & 31;
  const int lcol  = lane & 15;
  const int rbase = (lane >> 4) << 3;
#pragma unroll
  for (int s = 0; s < 4; ++s) {
    const int c    = ct * 64 + s * 16 + lcol;
    const float bb = bo[c];
    for (int j = 0; j < 8; ++j) {
      const int n = mt * 16 + rbase + j;
      const size_t idx = ((size_t)b * N_ + n) * D_ + c;
      out[idx] = acc[s][j] + bb + h[idx];
    }
  }
}

extern "C" void kernel_launch(void* const* d_in, const int* in_sizes, int n_in,
                              void* d_out, int out_size, void* d_ws, size_t ws_size,
                              hipStream_t stream) {
  const float* h  = (const float*)d_in[0];
  const int* mask = (const int*)d_in[1];
  const int* nc   = (const int*)d_in[2];
  const float* Wq = (const float*)d_in[3];
  const float* bq = (const float*)d_in[4];
  const float* Wk = (const float*)d_in[5];
  const float* bk = (const float*)d_in[6];
  const float* Wv = (const float*)d_in[7];
  const float* bv = (const float*)d_in[8];
  const float* Wo = (const float*)d_in[9];
  const float* bo = (const float*)d_in[10];
  float* out = (float*)d_out;
  (void)in_sizes; (void)n_in; (void)out_size; (void)ws_size;

  char* ws = (char*)d_ws;
  size_t off = 0;
  _Float16* h16 = (_Float16*)(ws + off); off += (size_t)B_ * N_ * D_ * 2;        // 16 MB
  _Float16* wt  = (_Float16*)(ws + off); off += (size_t)4 * D_ * D_ * 2;         // 512 KB
  _Float16* q16 = (_Float16*)(ws + off); off += (size_t)B_ * H_ * N_ * DK_ * 2;  // 16 MB
  _Float16* k16 = (_Float16*)(ws + off); off += (size_t)B_ * H_ * N_ * DK_ * 2;  // 16 MB
  _Float16* vt16= (_Float16*)(ws + off); off += (size_t)B_ * H_ * DK_ * N_ * 2;  // 16 MB
  _Float16* att = (_Float16*)(ws + off); off += (size_t)B_ * N_ * D_ * 2;        // 16 MB
  unsigned char* inv = (unsigned char*)(ws + off); off += (size_t)B_ * N_;       // 32 KB

  prep_h_kernel<<<(B_ * N_ * D_) / 256, 256, 0, stream>>>(h, h16, B_ * N_ * D_);
  prep_w_kernel<<<(4 * D_ * D_) / 256, 256, 0, stream>>>(Wq, Wk, Wv, Wo, wt);
  prep_inv_kernel<<<(B_ * N_) / 256, 256, 0, stream>>>(mask, nc, inv);

  qkv_kernel<<<dim3(B_ * (N_ / 16), D_ / 64, 3), 32, 0, stream>>>(
      h16, wt, bq, bk, bv, q16, k16, vt16);

  attn_kernel<<<B_ * H_ * (N_ / 16), 32, 0, stream>>>(q16, k16, vt16, inv, att);

  oproj_kernel<<<dim3(B_ * (N_ / 16), D_ / 64), 32, 0, stream>>>(att, wt, bo, h, out);
}
